// DGM_layer_59313498357972
// MI455X (gfx1250) — compile-verified, matrix operand order checked by reading, not run on hardware
//
#include <hip/hip_runtime.h>
#include <math.h>

typedef __attribute__((ext_vector_type(16))) _Float16 v16h;
typedef __attribute__((ext_vector_type(8)))  float    v8f;

// explicit global (device memory) address space -> global_load/store, not flat
#define AS1 __attribute__((address_space(1)))
typedef const AS1 _Float16* gh16_t;
typedef const AS1 v16h*     gv16_t;
typedef const AS1 float*    gf32_t;
typedef AS1 float*          gf32w_t;
typedef AS1 _Float16*       gh16w_t;

#define WMMA(a,b,c) __builtin_amdgcn_wmma_f32_16x16x32_f16(false,(a),false,(b),(short)0,(c),false,false)

#define MCC     32
#define SIGC    0.2f
#define MUC     0.05f
#define RC      0.05f
#define DELTAC  0.01f

// ---------------- workspace layout (bytes) ----------------
#define WS_L      0         // 20x20 f32 cholesky
#define WS_FPRIME 2048      // 2048*21 f32
#define WS_PART   174080    // 2048 f32
#define WS_PACK   182272    // f16 packed weight fragments
// pack offsets in f16 elements from WS_PACK
#define P_W1   0
#define P_U0   4096        // + g*4096   (g = z,g,r,h)
#define P_W0   20480       // + g*16384
#define P_WT0  86016       // + g*16384  (W transposed)
#define P_UT0  151552      // + g*4096   (U transposed, N padded to 32)
#define P_W1T  167936

struct Params {
  const float *inputs, *eps, *w1, *b1;
  const float *uzl, *wzl, *bzl, *ugl, *wgl, *bgl;
  const float *url, *wrl, *brl, *uhl, *whl, *bhl;
  const float *w, *b;
  float* out;
  char*  ws;
};

// ---------------- shared memory (65,312 B < 64 KB) ----------------
struct SM {
  float    dS[16*128];        // 8 KB  f32 gradient wrt hidden state
  float    dX[16*32];         // 2 KB  f32 gradient wrt input (padded to 32)
  _Float16 xproj[4][16*128];  // 16 KB x @ U_{z,g,r,h} (hoisted: weights shared across layers)
  _Float16 sstate[4][16*128]; // 16 KB s1, s_out0, s_out1, s_out2
  _Float16 gate[4][16*128];   // 16 KB z,g,r,h  (reused as da* in backward)
  _Float16 sr[16*128];        // 4 KB  s*r  (reused as dsr in backward)
  _Float16 xt[16*32];         // 1 KB  input tile, K-padded with zeros
  _Float16 viol[16*21];       // violation tile (interior kernel)
  float    valv[16];
  float    acc[16];
};

// ---- A-fragment (16x32 f16, MxK) from row-major f16 LDS array ----
__device__ inline v16h afrag(const _Float16* s, int pitch, int lane, int k0) {
  int m = lane & 15, half = lane >> 4;
  union { v16h v; unsigned int u[8]; } f;
#pragma unroll
  for (int v = 0; v < 8; ++v) {
    int k = k0 + ((v >> 2) << 4) + 8 * half + ((v & 3) << 1);
    f.u[v] = *(const unsigned int*)(s + m * pitch + k);   // ds_load_b32: two f16 (k, k+1)
  }
  return f.v;
}

// ---- C/D f32 fragment <-> row-major f32 LDS ----
__device__ inline v8f load_f32frag(const float* src, int pitch, int t, int lane) {
  int nlo = lane & 15, half = lane >> 4, n = 16 * t + nlo;
  v8f c;
#pragma unroll
  for (int v = 0; v < 8; ++v) c[v] = src[(v + 8 * half) * pitch + n];
  return c;
}
__device__ inline void store_f32frag(float* dst, int pitch, int t, int lane, v8f c) {
  int nlo = lane & 15, half = lane >> 4, n = 16 * t + nlo;
#pragma unroll
  for (int v = 0; v < 8; ++v) dst[(v + 8 * half) * pitch + n] = c[v];
}

// ---- GEMM over 8 N-tiles: out = act( [xp] + [bias] + sum_kk A_kk x B_kk ) ----
__device__ inline void gemm_store(const v16h* a, int nkk, gv16_t B,
                                  const _Float16* xp, gf32_t bias,
                                  _Float16* out, int lane, int do_tanh) {
  int nlo = lane & 15, half = lane >> 4;
  for (int t = 0; t < 8; ++t) {
    int n = 16 * t + nlo;
    v8f c;
#pragma unroll
    for (int v = 0; v < 8; ++v) {
      float x = bias ? bias[n] : 0.f;
      if (xp) x += (float)xp[(v + 8 * half) * 128 + n];
      c[v] = x;
    }
#pragma unroll
    for (int kk = 0; kk < 4; ++kk) {
      if (kk < nkk) c = WMMA(a[kk], B[(t * nkk + kk) * 32 + lane], c);
    }
#pragma unroll
    for (int v = 0; v < 8; ++v) {
      float x = c[v];
      if (do_tanh) x = tanhf(x);
      out[(v + 8 * half) * 128 + n] = (_Float16)x;
    }
  }
}

// ---- forward gates of one layer: z,g,r,h into sm.gate[0..3], sr into sm.sr ----
__device__ void layer_gates(SM& sm, gh16_t pb, gf32_t const* gb,
                            const _Float16* sin, int lane) {
  v16h a[4];
#pragma unroll
  for (int kk = 0; kk < 4; ++kk) a[kk] = afrag(sin, 128, lane, 32 * kk);
  for (int g = 0; g < 3; ++g)   // z, g, r
    gemm_store(a, 4, (gv16_t)(pb + P_W0 + g * 16384), sm.xproj[g], gb[g], sm.gate[g], lane, 1);
  for (int idx = lane; idx < 2048; idx += 32)
    sm.sr[idx] = (_Float16)((float)sin[idx] * (float)sm.gate[2][idx]);
  v16h as[4];
#pragma unroll
  for (int kk = 0; kk < 4; ++kk) as[kk] = afrag(sm.sr, 128, lane, 32 * kk);
  gemm_store(as, 4, (gv16_t)(pb + P_W0 + 3 * 16384), sm.xproj[3], gb[3], sm.gate[3], lane, 1);
}

// ---- full network forward (+ optional input-gradient backward) on one 16-pt tile ----
__device__ void eval_net(SM& sm, const Params& P, int lane, bool bwd) {
  gh16_t pb = (gh16_t)(const _Float16*)(P.ws + WS_PACK);
  gf32_t gb[4] = { (gf32_t)P.bzl, (gf32_t)P.bgl, (gf32_t)P.brl, (gf32_t)P.bhl };
  gf32_t gw = (gf32_t)P.w;
  gf32_t gbias = (gf32_t)P.b;
  __builtin_prefetch(P.ws + WS_PACK, 0, 3);   // global_prefetch of weight packs

  v16h xa[1];
  xa[0] = afrag(sm.xt, 32, lane, 0);

  // x-projections (shared across all 3 layers) and s1
  for (int g = 0; g < 4; ++g)
    gemm_store(xa, 1, (gv16_t)(pb + P_U0 + g * 4096), nullptr, nullptr, sm.xproj[g], lane, 0);
  gemm_store(xa, 1, (gv16_t)(pb + P_W1), nullptr, (gf32_t)P.b1, sm.sstate[0], lane, 1);

  // recurrent layers
  for (int L = 0; L < 3; ++L) {
    const _Float16* sin = sm.sstate[L];
    layer_gates(sm, pb, gb, sin, lane);
    _Float16* sout = sm.sstate[L + 1];
    for (int idx = lane; idx < 2048; idx += 32) {
      float z = (float)sm.gate[0][idx], g = (float)sm.gate[1][idx];
      float h = (float)sm.gate[3][idx], s = (float)sin[idx];
      sout[idx] = (_Float16)((1.f - g) * h + z * s);
    }
  }

  // value head: val = s3 @ w + b
  if (lane < 16) {
    float acc = gbias[0];
    for (int n = 0; n < 128; ++n) acc += (float)sm.sstate[3][lane * 128 + n] * gw[n];
    sm.valv[lane] = acc;
  }
  if (!bwd) return;

  // ----- backward: d val / d x -----
  for (int idx = lane; idx < 2048; idx += 32) sm.dS[idx] = gw[idx & 127];
  for (int p = lane; p < 512; p += 32) sm.dX[p] = 0.f;

  for (int L = 2; L >= 0; --L) {
    const _Float16* sin = sm.sstate[L];
    layer_gates(sm, pb, gb, sin, lane);        // recompute gates of this layer

    // pass 1: daz, dag, dah, and dS := dS*z
    for (int idx = lane; idx < 2048; idx += 32) {
      float d = sm.dS[idx], si = (float)sin[idx];
      float zv = (float)sm.gate[0][idx], gv = (float)sm.gate[1][idx], hv = (float)sm.gate[3][idx];
      sm.gate[0][idx] = (_Float16)(d * si * (1.f - zv * zv));
      sm.gate[1][idx] = (_Float16)(-d * hv * (1.f - gv * gv));
      sm.gate[3][idx] = (_Float16)(d * (1.f - gv) * (1.f - hv * hv));
      sm.dS[idx] = d * zv;
    }
    // dsr = dah @ Wh^T
    {
      v16h a[4];
#pragma unroll
      for (int kk = 0; kk < 4; ++kk) a[kk] = afrag(sm.gate[3], 128, lane, 32 * kk);
      gemm_store(a, 4, (gv16_t)(pb + P_WT0 + 3 * 16384), nullptr, nullptr, sm.sr, lane, 0);
    }
    // pass 2: dS += dsr*r ; dar = dsr*s*(1-r^2)
    for (int idx = lane; idx < 2048; idx += 32) {
      float dsr = (float)sm.sr[idx], rv = (float)sm.gate[2][idx], si = (float)sin[idx];
      sm.dS[idx] += dsr * rv;
      sm.gate[2][idx] = (_Float16)(dsr * si * (1.f - rv * rv));
    }
    // dS += daz@Wz^T + dag@Wg^T + dar@Wr^T ;  dX += da*@U*^T (incl. dah@Uh^T)
    v16h az[4], ag[4], ar[4], ah[4];
#pragma unroll
    for (int kk = 0; kk < 4; ++kk) {
      az[kk] = afrag(sm.gate[0], 128, lane, 32 * kk);
      ag[kk] = afrag(sm.gate[1], 128, lane, 32 * kk);
      ar[kk] = afrag(sm.gate[2], 128, lane, 32 * kk);
      ah[kk] = afrag(sm.gate[3], 128, lane, 32 * kk);
    }
    gv16_t wT0 = (gv16_t)(pb + P_WT0);
    gv16_t wT1 = (gv16_t)(pb + P_WT0 + 16384);
    gv16_t wT2 = (gv16_t)(pb + P_WT0 + 2 * 16384);
    for (int t = 0; t < 8; ++t) {
      v8f c = load_f32frag(sm.dS, 128, t, lane);
#pragma unroll
      for (int kk = 0; kk < 4; ++kk) c = WMMA(az[kk], wT0[(t * 4 + kk) * 32 + lane], c);
#pragma unroll
      for (int kk = 0; kk < 4; ++kk) c = WMMA(ag[kk], wT1[(t * 4 + kk) * 32 + lane], c);
#pragma unroll
      for (int kk = 0; kk < 4; ++kk) c = WMMA(ar[kk], wT2[(t * 4 + kk) * 32 + lane], c);
      store_f32frag(sm.dS, 128, t, lane, c);
    }
    gv16_t uT0 = (gv16_t)(pb + P_UT0);
    gv16_t uT1 = (gv16_t)(pb + P_UT0 + 4096);
    gv16_t uT2 = (gv16_t)(pb + P_UT0 + 2 * 4096);
    gv16_t uT3 = (gv16_t)(pb + P_UT0 + 3 * 4096);
    for (int t = 0; t < 2; ++t) {
      v8f c = load_f32frag(sm.dX, 32, t, lane);
#pragma unroll
      for (int kk = 0; kk < 4; ++kk) c = WMMA(az[kk], uT0[(t * 4 + kk) * 32 + lane], c);
#pragma unroll
      for (int kk = 0; kk < 4; ++kk) c = WMMA(ag[kk], uT1[(t * 4 + kk) * 32 + lane], c);
#pragma unroll
      for (int kk = 0; kk < 4; ++kk) c = WMMA(ar[kk], uT2[(t * 4 + kk) * 32 + lane], c);
#pragma unroll
      for (int kk = 0; kk < 4; ++kk) c = WMMA(ah[kk], uT3[(t * 4 + kk) * 32 + lane], c);
      store_f32frag(sm.dX, 32, t, lane, c);
    }
  }
  // input layer: dX += (dS*(1-s1^2)) @ w1^T
  for (int idx = lane; idx < 2048; idx += 32) {
    float d = sm.dS[idx], s1 = (float)sm.sstate[0][idx];
    sm.gate[0][idx] = (_Float16)(d * (1.f - s1 * s1));
  }
  {
    v16h a[4];
#pragma unroll
    for (int kk = 0; kk < 4; ++kk) a[kk] = afrag(sm.gate[0], 128, lane, 32 * kk);
    gv16_t w1T = (gv16_t)(pb + P_W1T);
    for (int t = 0; t < 2; ++t) {
      v8f c = load_f32frag(sm.dX, 32, t, lane);
#pragma unroll
      for (int kk = 0; kk < 4; ++kk) c = WMMA(a[kk], w1T[(t * 4 + kk) * 32 + lane], c);
      store_f32frag(sm.dX, 32, t, lane, c);
    }
  }
}

// ---------------- kernel 1: cholesky + weight packing ----------------
struct PD { const float* src; int sr, sc, Kp, Np, tr, dst; };

__global__ void k_prep(Params P) {
  gh16w_t pb = (gh16w_t)(_Float16*)(P.ws + WS_PACK);
  gf32w_t Lm = (gf32w_t)(float*)(P.ws + WS_L);
  if (blockIdx.x == 0 && threadIdx.x == 0) {
    float C[20][20];
    for (int i = 0; i < 20; ++i)
      for (int j = 0; j < 20; ++j) C[i][j] = 0.01f * ((i == j) ? 1.f : 0.5f);
    for (int j = 0; j < 20; ++j) {
      float s = C[j][j];
      for (int k = 0; k < j; ++k) s -= C[j][k] * C[j][k];
      C[j][j] = sqrtf(s);
      for (int i = j + 1; i < 20; ++i) {
        float t = C[i][j];
        for (int k = 0; k < j; ++k) t -= C[i][k] * C[j][k];
        C[i][j] = t / C[j][j];
      }
    }
    for (int i = 0; i < 20; ++i)
      for (int j = 0; j < 20; ++j) Lm[i * 20 + j] = (j <= i) ? C[i][j] : 0.f;
  }
  const PD pd[18] = {
    { P.w1, 21,128, 32,128,0, P_W1 },
    { P.uzl,21,128, 32,128,0, P_U0 + 0*4096 }, { P.ugl,21,128, 32,128,0, P_U0 + 1*4096 },
    { P.url,21,128, 32,128,0, P_U0 + 2*4096 }, { P.uhl,21,128, 32,128,0, P_U0 + 3*4096 },
    { P.wzl,128,128,128,128,0, P_W0 + 0*16384 }, { P.wgl,128,128,128,128,0, P_W0 + 1*16384 },
    { P.wrl,128,128,128,128,0, P_W0 + 2*16384 }, { P.whl,128,128,128,128,0, P_W0 + 3*16384 },
    { P.wzl,128,128,128,128,1, P_WT0 + 0*16384 }, { P.wgl,128,128,128,128,1, P_WT0 + 1*16384 },
    { P.wrl,128,128,128,128,1, P_WT0 + 2*16384 }, { P.whl,128,128,128,128,1, P_WT0 + 3*16384 },
    { P.uzl,21,128,128, 32,1, P_UT0 + 0*4096 }, { P.ugl,21,128,128, 32,1, P_UT0 + 1*4096 },
    { P.url,21,128,128, 32,1, P_UT0 + 2*4096 }, { P.uhl,21,128,128, 32,1, P_UT0 + 3*4096 },
    { P.w1, 21,128,128, 32,1, P_W1T },
  };
  for (int m = 0; m < 18; ++m) {
    int total = pd[m].Kp * pd[m].Np;
    int nkk = pd[m].Kp / 32;
    gf32_t src = (gf32_t)pd[m].src;
    for (int p = blockIdx.x * blockDim.x + threadIdx.x; p < total; p += gridDim.x * blockDim.x) {
      int e = p & 15, lane = (p >> 4) & 31, kt = p >> 9;
      int kk = kt % nkk, t = kt / nkk;
      int n = 16 * t + (lane & 15), half = lane >> 4, v = e >> 1;
      int k = 32 * kk + ((v >> 2) << 4) + 8 * half + ((v & 3) << 1) + (e & 1);
      float val = 0.f;
      if (pd[m].tr) { if (n < pd[m].sr && k < pd[m].sc) val = src[n * pd[m].sc + k]; }
      else          { if (k < pd[m].sr && n < pd[m].sc) val = src[k * pd[m].sc + n]; }
      pb[pd[m].dst + p] = (_Float16)val;
    }
  }
}

// ---------------- kernel 2: x1 (grad) and x2 (value + payoff) ----------------
__global__ __launch_bounds__(32) void k_main(Params P) {
  __shared__ SM sm;
  int lane = threadIdx.x, blk = blockIdx.x;
  gf32_t inp = (gf32_t)P.inputs;
  gf32w_t outp = (gf32w_t)P.out;
  if (blk < 128) {                       // x1 tiles
    int n0 = blk * 16;
    for (int p = lane; p < 512; p += 32) {
      int i = p >> 5, c = p & 31;
      sm.xt[p] = (_Float16)((c < 21) ? inp[(n0 + i) * 21 + c] : 0.f);
    }
    eval_net(sm, P, lane, true);
    gf32w_t fpr  = (gf32w_t)(float*)(P.ws + WS_FPRIME);
    gf32w_t part = (gf32w_t)(float*)(P.ws + WS_PART);
    for (int p = lane; p < 336; p += 32) {
      int i = p / 21, k = p % 21;
      fpr[(n0 + i) * 21 + k] = sm.dX[i * 32 + k];
    }
    if (lane < 16) {
      int i = lane;
      float t11 = sm.dX[i * 32 + 20];                   // drift last col = 1
      for (int k = 0; k < 20; ++k) t11 += MUC * inp[(n0 + i) * 21 + k] * sm.dX[i * 32 + k];
      part[n0 + i] = t11 - RC * sm.valv[i];
    }
  } else {                               // x2 tiles
    int r0 = 2048 + (blk - 128) * 16;
    for (int p = lane; p < 512; p += 32) {
      int i = p >> 5, c = p & 31;
      sm.xt[p] = (_Float16)((c < 21) ? inp[(r0 + i) * 21 + c] : 0.f);
    }
    eval_net(sm, P, lane, false);
    if (lane < 16) {
      int i = lane;
      float sl = 0.f;
      for (int k = 0; k < 20; ++k) sl += logf(inp[(r0 + i) * 21 + k]);
      float pay = fmaxf(0.f, expf(sl * (1.f / 20.f)));
      outp[r0 + i] = sm.valv[i] - pay;
    }
  }
}

// ---------------- kernel 3: interior term with MC loop (deterministic) ----------------
__global__ __launch_bounds__(32) void k_interior(Params P) {
  __shared__ SM sm;
  int lane = threadIdx.x, n0 = blockIdx.x * 16;
  gf32_t inp = (gf32_t)P.inputs;
  gf32_t epsp = (gf32_t)P.eps;
  gf32w_t outp = (gf32w_t)P.out;
  gf32_t Lm   = (gf32_t)(const float*)(P.ws + WS_L);
  gf32_t fpr  = (gf32_t)(const float*)(P.ws + WS_FPRIME);
  gf32_t part = (gf32_t)(const float*)(P.ws + WS_PART);
  if (lane < 16) sm.acc[lane] = 0.f;
  for (int p = lane; p < 512; p += 32) {   // static part of xt: time col + zero pad
    int i = p >> 5, c = p & 31;
    sm.xt[p] = (_Float16)((c == 20) ? inp[(n0 + i) * 21 + 20] : 0.f);
  }
  for (int mc = 0; mc < MCC; ++mc) {
    for (int p = lane; p < 320; p += 32) {
      int i = p / 20, k = p % 20;
      float s = 0.f;
      for (int d = 0; d < 20; ++d) s += epsp[((mc * 2048) + (n0 + i)) * 20 + d] * Lm[k * 20 + d];
      float loc = inp[(n0 + i) * 21 + k];
      float v = s * SIGC * loc;
      sm.viol[i * 21 + k] = (_Float16)v;
      sm.xt[i * 32 + k] = (_Float16)(loc + v);
    }
    eval_net(sm, P, lane, true);
    if (lane < 16) {
      int i = lane;
      float dot = 0.f;
      for (int k = 0; k < 20; ++k)
        dot += (sm.dX[i * 32 + k] - fpr[(n0 + i) * 21 + k]) * (float)sm.viol[i * 21 + k];
      sm.acc[i] += dot;
    }
  }
  if (lane < 16)
    outp[n0 + lane] = part[n0 + lane] + 0.5f * sm.acc[lane] / (MCC * DELTAC);
}

extern "C" void kernel_launch(void* const* d_in, const int* in_sizes, int n_in,
                              void* d_out, int out_size, void* d_ws, size_t ws_size,
                              hipStream_t stream) {
  (void)in_sizes; (void)n_in; (void)out_size; (void)ws_size;
  Params P;
  P.inputs = (const float*)d_in[0];  P.eps = (const float*)d_in[1];
  P.w1 = (const float*)d_in[2];      P.b1 = (const float*)d_in[3];
  P.uzl = (const float*)d_in[4];     P.wzl = (const float*)d_in[5];  P.bzl = (const float*)d_in[6];
  P.ugl = (const float*)d_in[7];     P.wgl = (const float*)d_in[8];  P.bgl = (const float*)d_in[9];
  P.url = (const float*)d_in[10];    P.wrl = (const float*)d_in[11]; P.brl = (const float*)d_in[12];
  P.uhl = (const float*)d_in[13];    P.whl = (const float*)d_in[14]; P.bhl = (const float*)d_in[15];
  P.w = (const float*)d_in[16];      P.b = (const float*)d_in[17];
  P.out = (float*)d_out;             P.ws = (char*)d_ws;

  k_prep<<<32, 256, 0, stream>>>(P);          // cholesky + f16 fragment packing
  k_main<<<256, 32, 0, stream>>>(P);          // x1 grads -> ws, x2 -> out[2048:]
  k_interior<<<128, 32, 0, stream>>>(P);      // MC interior -> out[:2048]
}
